// MultiHeadAttention2_49478023250155
// MI455X (gfx1250) — matrix operand order repaired and checked by reading, not compiled, not run on hardware
//
#include <hip/hip_runtime.h>
#include <hip/hip_bf16.h>

// ---------------------------------------------------------------------------
// MI455X (gfx1250) multi-head attention, f16 WMMA path with f32 accumulation.
// B=4, S=2048, D=1024, H=16, dk=64.
// ---------------------------------------------------------------------------

typedef __attribute__((ext_vector_type(16))) _Float16 v16h;
typedef __attribute__((ext_vector_type(8)))  _Float16 v8h;
typedef __attribute__((ext_vector_type(8)))  float    v8f;

#define D_MODEL 1024
#define SEQ     2048
#define NHEADS  16
#define DK      64
#define BS_ROWS (4 * SEQ)   // 8192

// ---- CDNA5 async global->LDS copy (ASYNCcnt path) --------------------------
// Generic LDS pointer low 32 bits == LDS byte address (ISA 10.2 aperture rule).

__device__ __forceinline__ void async_copy_b128(void* lds_dst,
                                                const void* gsrc) {
  unsigned lds = (unsigned)(unsigned long long)(size_t)lds_dst;
  unsigned long long ga = (unsigned long long)(size_t)gsrc;
  asm volatile("global_load_async_to_lds_b128 %0, %1, off"
               :: "v"(lds), "v"(ga) : "memory");
}

__device__ __forceinline__ void async_wait0() {
  asm volatile("s_wait_asynccnt 0x0" ::: "memory");
}

// ---- WMMA fragment loaders (layouts per CDNA5 ISA 7.12.2, wave32) ----------

// A-matrix 16x32 f16: lane holds row (lane&15); K chunks at (lane>>4)*8 and
// 16+(lane>>4)*8, each 8 contiguous halfwords.
__device__ __forceinline__ v16h frag_a(const _Float16* base, int ld, int lane) {
  int r  = lane & 15;
  int k0 = (lane >> 4) * 8;
  union { v16h v; v8h h[2]; } u;
  u.h[0] = *(const v8h*)(base + r * ld + k0);
  u.h[1] = *(const v8h*)(base + r * ld + k0 + 16);
  return u.v;
}

// B-matrix 32x16 f16 from TRANSPOSED storage baseT[col][k] (ld = ldt):
// lane holds column (lane&15); K chunks at (lane>>4)*8 and 16+(lane>>4)*8.
__device__ __forceinline__ v16h frag_b(const _Float16* baseT, int ldt, int lane) {
  int c  = lane & 15;
  int k0 = (lane >> 4) * 8;
  union { v16h v; v8h h[2]; } u;
  u.h[0] = *(const v8h*)(baseT + c * ldt + k0);
  u.h[1] = *(const v8h*)(baseT + c * ldt + k0 + 16);
  return u.v;
}

__device__ __forceinline__ v8f wmma_f16(v16h a, v16h b, v8f c) {
  return __builtin_amdgcn_wmma_f32_16x16x32_f16(false, a, false, b,
                                                (short)0, c, false, false);
}

// ---- f32 -> f16 conversion -------------------------------------------------

__global__ void cvt_f32_to_f16(const float* __restrict__ src,
                               _Float16* __restrict__ dst, int n) {
  int i = (blockIdx.x * blockDim.x + threadIdx.x) * 4;
  if (i + 3 < n) {
    float4 f = *(const float4*)(src + i);
    dst[i + 0] = (_Float16)f.x;
    dst[i + 1] = (_Float16)f.y;
    dst[i + 2] = (_Float16)f.z;
    dst[i + 3] = (_Float16)f.w;
  }
}

// ---- Tiled WMMA GEMM: C[M,N] = A[M,K] * B[K,N] + bias ----------------------
// 128x128 workgroup tile, 8 waves (4 M-groups x 2 N-groups), each wave 2x4
// 16x16 accumulators. K-step 32 staged through LDS (A via async-to-LDS,
// B transposed element-wise). Output type templated (f16 or f32).

template <typename OutT>
__global__ __launch_bounds__(256) void gemm_wmma(
    const _Float16* __restrict__ A, const _Float16* __restrict__ Bm,
    const float* __restrict__ bias, OutT* __restrict__ C,
    int M, int N, int Kd) {
  __shared__ _Float16 As[128 * 32];   // row-major, ld=32
  __shared__ _Float16 Bt[128 * 32];   // transposed: Bt[n][k], ld=32

  const int tid  = threadIdx.x;
  const int lane = tid & 31;
  const int w    = tid >> 5;
  const int wmg  = w & 3;     // M group: 32 rows
  const int wng  = w >> 2;    // N group: 64 cols
  const int m0   = blockIdx.y * 128;
  const int n0   = blockIdx.x * 128;

  // per-thread staging coordinates
  const int arow = tid >> 1, aseg = (tid & 1) * 16;           // A tile
  const int bkr  = tid >> 3, bnseg = (tid & 7) * 16;          // B tile

  v8f acc[2][4] = {};

  for (int kk = 0; kk < Kd; kk += 32) {
    __syncthreads();
    {   // stage A tile 128x32 via async global->LDS (2x b128 per thread)
      const _Float16* g = A + (size_t)(m0 + arow) * Kd + kk + aseg;
      async_copy_b128(As + arow * 32 + aseg, g);
      async_copy_b128(As + arow * 32 + aseg + 8, g + 8);
    }
    {   // stage B tile 32x128, transposed into Bt[n][k]
      const _Float16* g = Bm + (size_t)(kk + bkr) * N + n0 + bnseg;
#pragma unroll
      for (int i = 0; i < 16; ++i) Bt[(bnseg + i) * 32 + bkr] = g[i];
    }
    if (kk + 32 < Kd) {   // prefetch next K-step tiles into cache
      __builtin_prefetch(A + (size_t)(m0 + arow) * Kd + kk + 32 + aseg, 0, 1);
      __builtin_prefetch(Bm + (size_t)(kk + 32 + bkr) * N + n0 + bnseg, 0, 1);
    }
    async_wait0();
    __syncthreads();

    v16h a[2];
#pragma unroll
    for (int tm = 0; tm < 2; ++tm)
      a[tm] = frag_a(As + (wmg * 32 + tm * 16) * 32, 32, lane);
#pragma unroll
    for (int tn = 0; tn < 4; ++tn) {
      v16h bfr = frag_b(Bt + (wng * 64 + tn * 16) * 32, 32, lane);
#pragma unroll
      for (int tm = 0; tm < 2; ++tm)
        acc[tm][tn] = wmma_f16(a[tm], bfr, acc[tm][tn]);
    }
  }

  const int hi = (lane >> 4) * 8;
  const int nl = lane & 15;
#pragma unroll
  for (int tm = 0; tm < 2; ++tm) {
#pragma unroll
    for (int tn = 0; tn < 4; ++tn) {
      int col  = n0 + wng * 64 + tn * 16 + nl;
      float bv = bias[col];
      OutT* cp = C + (size_t)(m0 + wmg * 32 + tm * 16 + hi) * N + col;
#pragma unroll
      for (int v = 0; v < 8; ++v)
        cp[(size_t)v * N] = (OutT)(acc[tm][tn][v] + bv);
    }
  }
}

// ---- Flash attention: per (b,h,128-query block) ----------------------------
// Q/K/V stored as [B*S, 1024] with head h at column offset h*64.
// LDS: Qs(16K) Ks(16K) Vt(16K) Ss f32(64K) Ps f16(32K) stats(1.5K) = 148992 B.

__global__ __launch_bounds__(256) void attn_wmma(
    const _Float16* __restrict__ Q, const _Float16* __restrict__ K,
    const _Float16* __restrict__ V, _Float16* __restrict__ O) {
  extern __shared__ char smem[];
  _Float16* Qs = (_Float16*)(smem);            // 128x64, ld=64
  _Float16* Ks = (_Float16*)(smem + 16384);    // 128x64, ld=64 (= B^T for QK^T)
  _Float16* Vt = (_Float16*)(smem + 32768);    // 64x128, ld=128 (V transposed)
  float*    Ss = (float*)   (smem + 49152);    // 128x128 f32
  _Float16* Ps = (_Float16*)(smem + 114688);   // 128x128 f16
  float* mrow  = (float*)   (smem + 147456);   // 128
  float* lrow  = mrow + 128;
  float* srow  = lrow + 128;

  const int tid  = threadIdx.x;
  const int lane = tid & 31;
  const int w    = tid >> 5;
  const int wmg  = w & 3;    // 32 query rows
  const int wng  = w >> 2;   // scores: 64 cols; PV: 32 cols
  const int hi   = (lane >> 4) * 8;
  const int nl   = lane & 15;

  const int q0 = blockIdx.x * 128;
  const int bh = blockIdx.y;
  const int b  = bh >> 4, h = bh & 15;
  const size_t ldg  = D_MODEL;
  const size_t base = (size_t)b * SEQ * ldg + (size_t)h * DK;

  const int srow8 = tid >> 1;            // staging row (Q/K/V tiles)
  const int sseg  = (tid & 1) * 32;      // staging 32-halfword segment

  {   // stage Q block (128x64) via async global->LDS
    const _Float16* g = Q + base + (size_t)(q0 + srow8) * ldg + sseg;
#pragma unroll
    for (int i = 0; i < 4; ++i)
      async_copy_b128(Qs + srow8 * 64 + sseg + i * 8, g + i * 8);
  }
  if (tid < 128) { mrow[tid] = -3.0e38f; lrow[tid] = 0.f; }

  v8f o[2][2] = {};

  for (int kv = 0; kv < SEQ; kv += 128) {
    __syncthreads();
    {   // stage K block (128x64, row-major == transposed K^T operand)
      const _Float16* g = K + base + (size_t)(kv + srow8) * ldg + sseg;
#pragma unroll
      for (int i = 0; i < 4; ++i)
        async_copy_b128(Ks + srow8 * 64 + sseg + i * 8, g + i * 8);
    }
    {   // stage V block transposed: Vt[d][k]
      const _Float16* g = V + base + (size_t)(kv + srow8) * ldg + sseg;
#pragma unroll
      for (int i = 0; i < 32; ++i) Vt[(sseg + i) * 128 + srow8] = g[i];
    }
    if (kv + 128 < SEQ) {   // prefetch next K/V tiles into cache
      __builtin_prefetch(K + base + (size_t)(kv + 128 + srow8) * ldg + sseg, 0, 1);
      __builtin_prefetch(V + base + (size_t)(kv + 128 + srow8) * ldg + sseg, 0, 1);
    }
    async_wait0();
    __syncthreads();

    // ---- scores S = Q K^T * 1/sqrt(dk): 128x128, dk loop 2 x 32 ----
    v8f sacc[2][4] = {};
#pragma unroll
    for (int kk = 0; kk < DK; kk += 32) {
      v16h a[2];
#pragma unroll
      for (int tm = 0; tm < 2; ++tm)
        a[tm] = frag_a(Qs + (wmg * 32 + tm * 16) * 64 + kk, 64, lane);
#pragma unroll
      for (int tn = 0; tn < 4; ++tn) {
        v16h bfr = frag_b(Ks + (wng * 64 + tn * 16) * 64 + kk, 64, lane);
#pragma unroll
        for (int tm = 0; tm < 2; ++tm)
          sacc[tm][tn] = wmma_f16(a[tm], bfr, sacc[tm][tn]);
      }
    }
#pragma unroll
    for (int tm = 0; tm < 2; ++tm)
#pragma unroll
      for (int tn = 0; tn < 4; ++tn)
#pragma unroll
        for (int v = 0; v < 8; ++v)
          Ss[(wmg * 32 + tm * 16 + v + hi) * 128 + wng * 64 + tn * 16 + nl] =
              sacc[tm][tn][v] * 0.125f;
    __syncthreads();

    // ---- online softmax (one thread per query row) ----
    if (tid < 128) {
      float mold = mrow[tid];
      float mx   = mold;
      const float* sr = Ss + tid * 128;
      for (int j = 0; j < 128; ++j) mx = fmaxf(mx, sr[j]);
      float sc  = exp2f((mold - mx) * 1.44269504f);
      float sum = 0.f;
      _Float16* pr = Ps + tid * 128;
      for (int j = 0; j < 128; ++j) {
        float p = exp2f((sr[j] - mx) * 1.44269504f);
        pr[j]   = (_Float16)p;
        sum    += p;
      }
      lrow[tid] = lrow[tid] * sc + sum;
      mrow[tid] = mx;
      srow[tid] = sc;
    }
    __syncthreads();

    // ---- rescale O accumulators then O += P V ----
#pragma unroll
    for (int tm = 0; tm < 2; ++tm)
#pragma unroll
      for (int v = 0; v < 8; ++v) {
        float sc = srow[wmg * 32 + tm * 16 + v + hi];
        o[tm][0][v] = o[tm][0][v] * sc;
        o[tm][1][v] = o[tm][1][v] * sc;
      }
#pragma unroll
    for (int kk = 0; kk < 128; kk += 32) {
      v16h a[2];
#pragma unroll
      for (int tm = 0; tm < 2; ++tm)
        a[tm] = frag_a(Ps + (wmg * 32 + tm * 16) * 128 + kk, 128, lane);
#pragma unroll
      for (int tn = 0; tn < 2; ++tn) {
        v16h bfr = frag_b(Vt + (wng * 32 + tn * 16) * 128 + kk, 128, lane);
#pragma unroll
        for (int tm = 0; tm < 2; ++tm)
          o[tm][tn] = wmma_f16(a[tm], bfr, o[tm][tn]);
      }
    }
  }

  // ---- finalize: divide by row sums, store f16 heads-concatenated ----
#pragma unroll
  for (int tm = 0; tm < 2; ++tm)
#pragma unroll
    for (int tn = 0; tn < 2; ++tn)
#pragma unroll
      for (int v = 0; v < 8; ++v) {
        int r     = wmg * 32 + tm * 16 + v + hi;
        float inv = 1.f / lrow[r];
        size_t row = (size_t)(q0 + r);
        O[base + row * ldg + wng * 32 + tn * 16 + nl] =
            (_Float16)(o[tm][tn][v] * inv);
      }
}

// ---------------------------------------------------------------------------

extern "C" void kernel_launch(void* const* d_in, const int* in_sizes, int n_in,
                              void* d_out, int out_size, void* d_ws,
                              size_t ws_size, hipStream_t stream) {
  const float* x  = (const float*)d_in[0];
  const float* Wq = (const float*)d_in[1];
  const float* bq = (const float*)d_in[2];
  const float* Wk = (const float*)d_in[3];
  const float* bk = (const float*)d_in[4];
  const float* Wv = (const float*)d_in[5];
  const float* bv = (const float*)d_in[6];
  const float* Wo = (const float*)d_in[7];
  const float* bo = (const float*)d_in[8];
  float* out = (float*)d_out;

  // workspace layout (bytes); total ~92 MB
  char* ws = (char*)d_ws;
  _Float16* xh  = (_Float16*)(ws + 0);          // 16 MB
  _Float16* Wqh = (_Float16*)(ws + 16777216);   //  2 MB
  _Float16* Wkh = (_Float16*)(ws + 18874368);
  _Float16* Wvh = (_Float16*)(ws + 20971520);
  _Float16* Woh = (_Float16*)(ws + 23068672);
  _Float16* Qh  = (_Float16*)(ws + 25165824);   // 16 MB
  _Float16* Kh  = (_Float16*)(ws + 41943040);
  _Float16* Vh  = (_Float16*)(ws + 58720256);
  _Float16* Ah  = (_Float16*)(ws + 75497472);   // 16 MB

  const int nX = BS_ROWS * D_MODEL;   // 8388608
  const int nW = D_MODEL * D_MODEL;   // 1048576

  cvt_f32_to_f16<<<nX / 4 / 256, 256, 0, stream>>>(x, xh, nX);
  cvt_f32_to_f16<<<nW / 4 / 256, 256, 0, stream>>>(Wq, Wqh, nW);
  cvt_f32_to_f16<<<nW / 4 / 256, 256, 0, stream>>>(Wk, Wkh, nW);
  cvt_f32_to_f16<<<nW / 4 / 256, 256, 0, stream>>>(Wv, Wvh, nW);
  cvt_f32_to_f16<<<nW / 4 / 256, 256, 0, stream>>>(Wo, Woh, nW);

  dim3 gg(D_MODEL / 128, BS_ROWS / 128);  // (8, 64)
  gemm_wmma<_Float16><<<gg, 256, 0, stream>>>(xh, Wqh, bq, Qh,
                                              BS_ROWS, D_MODEL, D_MODEL);
  gemm_wmma<_Float16><<<gg, 256, 0, stream>>>(xh, Wkh, bk, Kh,
                                              BS_ROWS, D_MODEL, D_MODEL);
  gemm_wmma<_Float16><<<gg, 256, 0, stream>>>(xh, Wvh, bv, Vh,
                                              BS_ROWS, D_MODEL, D_MODEL);

  const int attn_smem = 148992;  // fits CDNA5's 320 KB/WGP LDS
  hipFuncSetAttribute((const void*)attn_wmma,
                      hipFuncAttributeMaxDynamicSharedMemorySize, attn_smem);
  attn_wmma<<<dim3(SEQ / 128, 4 * NHEADS), 256, attn_smem, stream>>>(Qh, Kh,
                                                                     Vh, Ah);

  gemm_wmma<float><<<gg, 256, 0, stream>>>(Ah, Woh, bo, out,
                                           BS_ROWS, D_MODEL, D_MODEL);
}
